// MHA_3925600108895
// MI455X (gfx1250) — matrix operand order repaired and checked by reading, not compiled, not run on hardware
//
#include <hip/hip_runtime.h>
#include <cstdint>

// ---------------------------------------------------------------------------
// Types for CDNA5 WMMA (wave32) + TDM descriptors.
// ---------------------------------------------------------------------------
typedef __bf16 bf16_t;
typedef bf16_t v16bf __attribute__((ext_vector_type(16)));
typedef bf16_t v2bf  __attribute__((ext_vector_type(2)));
typedef float  v8f   __attribute__((ext_vector_type(8)));
typedef unsigned v4u __attribute__((ext_vector_type(4)));
typedef int      v8i __attribute__((ext_vector_type(8)));
typedef int      v4i __attribute__((ext_vector_type(4)));

union Frag { unsigned u[8]; v16bf v; };

// Native bf16 convert (compiler picks v_cvt_*bf16* on gfx1250).
__device__ __forceinline__ unsigned pack2(float a, float b) {
  union { v2bf v; unsigned u; } c;
  c.v[0] = (bf16_t)a; c.v[1] = (bf16_t)b;
  return c.u;
}
// K order used here for A and B (contiguous, k = 16*hf + 2v): lane half selects K half, VGPR v holds a K pair.
__device__ __forceinline__ int kbA(int v, int hf) {
  return 2 * v + 16 * hf;
}

// ---------------------------------------------------------------------------
// Tensor Data Mover: 2D bf16 tile (tile_d0 contiguous elems x tile_d1 rows,
// row stride stride0 elems) from global -> LDS byte offset lds_off.
// D# per CDNA5 ISA. Groups 2/3 zero => dims 2..4 unused.
// This toolchain uses the 6-arg builtin: (v4u, v8i, v4i, v4i, v8i, i32 cpol).
// ---------------------------------------------------------------------------
__device__ __forceinline__ void tdm_load_2d(
    unsigned lds_off, const unsigned short* gptr,
    unsigned tile_d0, unsigned tile_d1,
    unsigned tensor_d0, unsigned tensor_d1, unsigned stride0)
{
  const unsigned long long ga = (unsigned long long)(uintptr_t)gptr;
  v4u g0;
  g0[0] = 1u;                                    // count=1, user mode, no gather
  g0[1] = lds_off;                               // lds_addr (bytes)
  g0[2] = (unsigned)(ga & 0xFFFFFFFFu);          // global_addr[31:0]
  g0[3] = (unsigned)((ga >> 32) & 0x01FFFFFFu)   // global_addr[56:32]
        | (2u << 30);                            // type = 2 ("image")
  v8i g1;
  g1[0] = (int)(1u << 16);                                   // data_size=1 (2B)
  g1[1] = (int)((tensor_d0 & 0xFFFFu) << 16);                // tensor_dim0 lo
  g1[2] = (int)((tensor_d0 >> 16) | ((tensor_d1 & 0xFFFFu) << 16));
  g1[3] = (int)((tensor_d1 >> 16) | (tile_d0 << 16));        // tile_dim0
  g1[4] = (int)(tile_d1 & 0xFFFFu);                          // tile_dim1, dim2=0
  g1[5] = (int)stride0;                                      // dim0 stride lo
  g1[6] = 0;                                                 // stride hi / s1 lo
  g1[7] = 0;
  const v4i z4 = {0, 0, 0, 0};
  const v8i z8 = {0, 0, 0, 0, 0, 0, 0, 0};
  __builtin_amdgcn_tensor_load_to_lds(g0, g1, z4, z4, z8, 0);
}

#define GM 4096
#define GN 1024
#define GK 1024

// ---------------------------------------------------------------------------
// Bulk f32 -> bf16 convert: one thread = 4 elements.
// ---------------------------------------------------------------------------
__global__ __launch_bounds__(256) void cvt_f32_to_bf16(
    const float* __restrict__ in, unsigned* __restrict__ out)
{
  const size_t t = (size_t)blockIdx.x * 256 + threadIdx.x;
  const float4 p = *(const float4*)(in + t * 4);
  out[t * 2 + 0] = pack2(p.x, p.y);
  out[t * 2 + 1] = pack2(p.z, p.w);
}

// ---------------------------------------------------------------------------
// GEMM: out = A[4096x1024](bf16) * W[1024x1024](bf16)^T + bias
// TDM double-buffered LDS staging; 8 waves; block tile 64m x 128n.
// mode 0: f32 [B,H,S,D]   mode 1: f32 row-major   mode 2: bf16 [B,H,S,D]
// ---------------------------------------------------------------------------
__global__ __launch_bounds__(256) void wmma_gemm_bias(
    const unsigned short* __restrict__ A, const unsigned short* __restrict__ W,
    const float* __restrict__ bias, void* outp, int mode)
{
  __shared__ unsigned short lx[2][64 * 32];    // A tiles, bf16 [m][k]
  __shared__ unsigned short lw[2][128 * 32];   // W tiles, bf16 [n][k]

  const int tid  = threadIdx.x;
  const int wave = tid >> 5, lane = tid & 31;
  const int hf   = lane >> 4, l15 = lane & 15;
  const int wm0  = (wave & 3) * 16;
  const int wn0  = (wave >> 2) * 64;
  const int bm0  = blockIdx.x * 64;
  const int bn0  = blockIdx.y * 128;
  const unsigned lxo = (unsigned)(uintptr_t)&lx[0][0];
  const unsigned lwo = (unsigned)(uintptr_t)&lw[0][0];

  v8f vzero = {0.f,0.f,0.f,0.f,0.f,0.f,0.f,0.f};
  v8f acc[4];
#pragma unroll
  for (int nt = 0; nt < 4; ++nt) acc[nt] = vzero;

  if (wave == 0) {   // prologue: DMA tile 0 into buffer 0
    tdm_load_2d(lxo, A + (size_t)bm0 * GK, 32, 64, GK, GM, GK);
    tdm_load_2d(lwo, W + (size_t)bn0 * GK, 32, 128, GK, GN, GK);
  }

  int cur = 0;
  for (int k0 = 0; k0 < GK; k0 += 32) {
    if (wave == 0) {
      if (k0 + 32 < GK) {            // DMA next tile, then wait for current
        const int nxt = cur ^ 1;
        tdm_load_2d(lxo + (unsigned)nxt * (64 * 32 * 2),
                    A + (size_t)bm0 * GK + (k0 + 32), 32, 64, GK, GM, GK);
        tdm_load_2d(lwo + (unsigned)nxt * (128 * 32 * 2),
                    W + (size_t)bn0 * GK + (k0 + 32), 32, 128, GK, GN, GK);
        __builtin_amdgcn_s_wait_tensorcnt(2);   // in-order: current tile done
      } else {
        __builtin_amdgcn_s_wait_tensorcnt(0);
      }
    }
    __syncthreads();

    Frag a;
#pragma unroll
    for (int v = 0; v < 8; ++v)
      a.u[v] = *(const unsigned*)&lx[cur][(wm0 + l15) * 32 + kbA(v, hf)];
#pragma unroll
    for (int nt = 0; nt < 4; ++nt) {
      Frag b;
      const int r = wn0 + nt * 16 + l15;
#pragma unroll
      for (int v = 0; v < 8; ++v)
        b.u[v] = *(const unsigned*)&lw[cur][r * 32 + 2 * v + 16 * hf];
      acc[nt] = __builtin_amdgcn_wmma_f32_16x16x32_bf16(
          false, a.v, false, b.v, (short)0, acc[nt], false, false);
    }
    __syncthreads();   // reads done before this buffer is DMA'd again
    cur ^= 1;
  }

  float* outf  = (float*)outp;
  bf16_t* outb = (bf16_t*)outp;
#pragma unroll
  for (int nt = 0; nt < 4; ++nt) {
    const int ncol = bn0 + wn0 + nt * 16 + l15;
    const float bv = bias[ncol];
#pragma unroll
    for (int i = 0; i < 8; ++i) {
      const int grow = bm0 + wm0 + i + 8 * hf;   // C layout: VGPR i -> M=i / i+8
      const float val = acc[nt][i] + bv;
      if (mode == 1) {
        outf[(size_t)grow * GN + ncol] = val;
      } else {
        const int bb = grow >> 11, ss = grow & 2047;
        const int hh = ncol >> 6,  dd = ncol & 63;
        const size_t idx = (((size_t)(bb * 16 + hh)) * 2048 + ss) * 64 + dd;
        if (mode == 0) outf[idx] = val;
        else           outb[idx] = (bf16_t)val;
      }
    }
  }
}

// ---------------------------------------------------------------------------
// RoPE: read f32 [B*H,S,64], write bf16 (packed pairs). One thread per pair.
// out[2i]   = x[2i]*cos(a(2i))     - x[2i+1]*sin(a(2i))
// out[2i+1] = x[2i+1]*cos(a(2i+1)) + x[2i]*sin(a(2i+1)), a(d)=s*10000^(-(d&31)/32)
// ---------------------------------------------------------------------------
__global__ __launch_bounds__(256) void rope_to_bf16(
    const float* __restrict__ in, unsigned* __restrict__ out)
{
  const int idx = blockIdx.x * 256 + threadIdx.x;   // 2M threads
  const int i   = idx & 31;
  const int row = idx >> 5;                          // bh*2048 + s
  const int s   = row & 2047;
  const size_t base = (size_t)row * 64;
  const int d0 = 2 * i, d1 = 2 * i + 1;
  const float fs = (float)s;
  const float a0 = fs * __powf(10000.f, -(float)(d0 & 31) * (1.f / 32.f));
  const float a1 = fs * __powf(10000.f, -(float)(d1 & 31) * (1.f / 32.f));
  const float x0 = in[base + d0], x1 = in[base + d1];
  float s0, c0, s1, c1;
  __sincosf(a0, &s0, &c0);
  __sincosf(a1, &s1, &c1);
  out[(size_t)row * 32 + i] = pack2(x0 * c0 - x1 * s0, x1 * c1 + x0 * s1);
}

// ---------------------------------------------------------------------------
// Flash attention: 4 waves/block, each wave one 16-row Q tile of the same
// (b,h). K/V tiles (32 keys x 64d bf16 = contiguous 4KB) DMA'd by TDM with
// double buffering; f32 online softmax; ctx written bf16 as [B,S,H,D].
// ---------------------------------------------------------------------------
__global__ __launch_bounds__(128) void flash_attn(
    const unsigned short* __restrict__ Q, const unsigned short* __restrict__ K,
    const unsigned short* __restrict__ V, unsigned short* __restrict__ ctx)
{
  __shared__ unsigned short kt[2][32 * 64];
  __shared__ unsigned short vt[2][32 * 64];
  __shared__ float pt[4][16 * 32];            // per-wave P scratch [m][key]

  const int tid  = threadIdx.x;
  const int wave = tid >> 5, lane = tid & 31;
  const int hf   = lane >> 4, l15 = lane & 15;
  const int qtile = blockIdx.x * 4 + wave;    // 4096 q-tiles
  const int bh = qtile >> 7;
  const int q0 = (qtile & 127) << 4;
  const unsigned short* Qb = Q + ((size_t)bh * 2048 + q0) * 64;
  const unsigned short* Kb = K + (size_t)bh * 2048 * 64;
  const unsigned short* Vb = V + (size_t)bh * 2048 * 64;
  const unsigned kto = (unsigned)(uintptr_t)&kt[0][0];
  const unsigned vto = (unsigned)(uintptr_t)&vt[0][0];

  Frag qa[2];                                 // Q as A: 16x64, two k-steps
#pragma unroll
  for (int f = 0; f < 2; ++f)
#pragma unroll
    for (int v = 0; v < 8; ++v) {
      const int k = kbA(v, hf) + 32 * f;
      qa[f].u[v] = *(const unsigned*)(Qb + (size_t)l15 * 64 + k);
    }

  float mstate[8], lstate[8];
  v8f vzero = {0.f,0.f,0.f,0.f,0.f,0.f,0.f,0.f};
  v8f acc[4];
#pragma unroll
  for (int i = 0; i < 8; ++i) { mstate[i] = -3.0e38f; lstate[i] = 0.f; }
#pragma unroll
  for (int nt = 0; nt < 4; ++nt) acc[nt] = vzero;

  if (wave == 0) {   // prologue DMA: first 32-key K and V tiles (4KB each)
    tdm_load_2d(kto, Kb, 2048, 1, 131072, 1, 2048);
    tdm_load_2d(vto, Vb, 2048, 1, 131072, 1, 2048);
  }

  int cur = 0;
  for (int kb = 0; kb < 2048; kb += 32) {
    if (wave == 0) {
      if (kb + 32 < 2048) {
        const int nxt = cur ^ 1;
        tdm_load_2d(kto + (unsigned)nxt * (32 * 64 * 2),
                    Kb + (size_t)(kb + 32) * 64, 2048, 1, 131072, 1, 2048);
        tdm_load_2d(vto + (unsigned)nxt * (32 * 64 * 2),
                    Vb + (size_t)(kb + 32) * 64, 2048, 1, 131072, 1, 2048);
        __builtin_amdgcn_s_wait_tensorcnt(2);
      } else {
        __builtin_amdgcn_s_wait_tensorcnt(0);
      }
    }
    __syncthreads();

    // S = Q * K^T : two 16-key tiles, d-dim in two steps of 32
    v8f sc[2] = {vzero, vzero};
#pragma unroll
    for (int t = 0; t < 2; ++t)
#pragma unroll
      for (int f = 0; f < 2; ++f) {
        Frag b;
#pragma unroll
        for (int v = 0; v < 8; ++v) {
          const int k = 2 * v + 16 * hf + 32 * f;
          b.u[v] = *(const unsigned*)&kt[cur][(t * 16 + l15) * 64 + k];
        }
        sc[t] = __builtin_amdgcn_wmma_f32_16x16x32_bf16(
            false, qa[f].v, false, b.v, (short)0, sc[t], false, false);
      }

    // online softmax per row (VGPR slot i -> rows i / i+8 in the two halves)
#pragma unroll
    for (int i = 0; i < 8; ++i) {
      const float s0 = sc[0][i] * 0.125f;   // 1/sqrt(64)
      const float s1 = sc[1][i] * 0.125f;
      float mloc = fmaxf(s0, s1);
#pragma unroll
      for (int m = 8; m >= 1; m >>= 1) mloc = fmaxf(mloc, __shfl_xor(mloc, m, 32));
      const float mnew  = fmaxf(mstate[i], mloc);
      const float alpha = __expf(mstate[i] - mnew);
      const float p0 = __expf(s0 - mnew);
      const float p1 = __expf(s1 - mnew);
      float rs = p0 + p1;
#pragma unroll
      for (int m = 8; m >= 1; m >>= 1) rs += __shfl_xor(rs, m, 32);
      lstate[i] = lstate[i] * alpha + rs;
      mstate[i] = mnew;
      acc[0][i] *= alpha; acc[1][i] *= alpha; acc[2][i] *= alpha; acc[3][i] *= alpha;
      pt[wave][(i + 8 * hf) * 32 + l15]      = p0;
      pt[wave][(i + 8 * hf) * 32 + 16 + l15] = p1;
    }
    __builtin_amdgcn_wave_barrier();   // same-wave LDS is in-order (DScnt)

    Frag pa;                            // P (16x32) as A fragment
#pragma unroll
    for (int v = 0; v < 8; ++v) {
      const int k = kbA(v, hf);
      pa.u[v] = pack2(pt[wave][l15 * 32 + k], pt[wave][l15 * 32 + k + 1]);
    }
    // O += P * V : K-dim = 32 keys, N = 64 d in 4 tiles
#pragma unroll
    for (int nt = 0; nt < 4; ++nt) {
      Frag vb;
#pragma unroll
      for (int v = 0; v < 8; ++v) {
        const int k = 2 * v + 16 * hf;
        const int d = nt * 16 + l15;
        const unsigned lo = vt[cur][k * 64 + d];
        const unsigned hi = vt[cur][(k + 1) * 64 + d];
        vb.u[v] = lo | (hi << 16);
      }
      acc[nt] = __builtin_amdgcn_wmma_f32_16x16x32_bf16(
          false, pa.v, false, vb.v, (short)0, acc[nt], false, false);
    }
    __syncthreads();
    cur ^= 1;
  }

  // normalize; write ctx bf16 as [B,S,H,D] so the final GEMM reads rows of E
  const int bb = bh >> 4, hh = bh & 15;
  bf16_t* cb = (bf16_t*)ctx;
#pragma unroll
  for (int i = 0; i < 8; ++i) {
    const float inv = 1.f / lstate[i];
    const int srow = q0 + i + 8 * hf;
#pragma unroll
    for (int nt = 0; nt < 4; ++nt)
      cb[(((size_t)bb * 2048 + srow) * 16 + hh) * 64 + nt * 16 + l15] =
          (bf16_t)(acc[nt][i] * inv);
  }
}

// ---------------------------------------------------------------------------
extern "C" void kernel_launch(void* const* d_in, const int* in_sizes, int n_in,
                              void* d_out, int out_size, void* d_ws, size_t ws_size,
                              hipStream_t stream) {
  const float* x  = (const float*)d_in[0];
  const float* Wq = (const float*)d_in[1];
  const float* bq = (const float*)d_in[2];
  const float* Wk = (const float*)d_in[3];
  const float* bk = (const float*)d_in[4];
  const float* Wv = (const float*)d_in[5];
  const float* bv = (const float*)d_in[6];
  const float* Wo = (const float*)d_in[7];
  const float* bo = (const float*)d_in[8];

  char* ws = (char*)d_ws;                       // 80 MB total
  float*          Qf    = (float*)(ws + 0);                  // f32 [B,H,S,D]
  float*          Kf    = (float*)(ws + (16u << 20));
  unsigned short* Qbf   = (unsigned short*)(ws + (32u << 20));
  unsigned short* Kbf   = (unsigned short*)(ws + (40u << 20));
  unsigned short* Vbf   = (unsigned short*)(ws + (48u << 20));
  unsigned short* xbf   = (unsigned short*)(ws + (56u << 20));
  unsigned short* Wqbf  = (unsigned short*)(ws + (64u << 20));
  unsigned short* Wkbf  = (unsigned short*)(ws + (66u << 20));
  unsigned short* Wvbf  = (unsigned short*)(ws + (68u << 20));
  unsigned short* Wobf  = (unsigned short*)(ws + (70u << 20));
  unsigned short* CTXbf = (unsigned short*)(ws + (72u << 20));

  cvt_f32_to_bf16<<<4096, 256, 0, stream>>>(x,  (unsigned*)xbf);
  cvt_f32_to_bf16<<<1024, 256, 0, stream>>>(Wq, (unsigned*)Wqbf);
  cvt_f32_to_bf16<<<1024, 256, 0, stream>>>(Wk, (unsigned*)Wkbf);
  cvt_f32_to_bf16<<<1024, 256, 0, stream>>>(Wv, (unsigned*)Wvbf);
  cvt_f32_to_bf16<<<1024, 256, 0, stream>>>(Wo, (unsigned*)Wobf);

  dim3 gg(GM / 64, GN / 128);
  wmma_gemm_bias<<<gg, 256, 0, stream>>>(xbf, Wqbf, bq, Qf,  0);
  wmma_gemm_bias<<<gg, 256, 0, stream>>>(xbf, Wkbf, bk, Kf,  0);
  wmma_gemm_bias<<<gg, 256, 0, stream>>>(xbf, Wvbf, bv, Vbf, 2);
  rope_to_bf16<<<8192, 256, 0, stream>>>(Qf, (unsigned*)Qbf);
  rope_to_bf16<<<8192, 256, 0, stream>>>(Kf, (unsigned*)Kbf);
  flash_attn<<<1024, 128, 0, stream>>>(Qbf, Kbf, Vbf, CTXbf);
  wmma_gemm_bias<<<gg, 256, 0, stream>>>(CTXbf, Wobf, bo, d_out, 1);
}
